// Seq2Seq_85358180041288
// MI455X (gfx1250) — matrix operand order repaired and checked by reading, not compiled, not run on hardware
//
#include <hip/hip_runtime.h>
#include <hip/hip_bf16.h>

typedef __bf16 bf16_t;
typedef __attribute__((ext_vector_type(16))) __bf16 v16bf;
typedef __attribute__((ext_vector_type(8)))  __bf16 v8bf;
typedef __attribute__((ext_vector_type(8)))  float  v8f;

// ---- model dims ----
constexpr int V  = 100000;
constexpr int H  = 300;
constexpr int L  = 64;
constexpr int B  = 64;
constexpr int H3 = 3 * H;          // 900
constexpr int KP = 320;            // K padded to multiple of 32
constexpr int KT = 10;             // k-tiles of 32
constexpr int NT_G   = 57;         // ceil(900/16) gate n-tiles
constexpr int NPAD_G = NT_G * 16;  // 912
constexpr int NT_V   = V / 16;     // 6250 vocab n-tiles (even -> uniform pairs)

// packed B-fragment tile: 32 lanes x 16 bf16 = 512 elems = 1024 bytes
constexpr size_t TILE_ELEMS = 512;
constexpr size_t SZ_OUTW = (size_t)NT_V * KT * TILE_ELEMS;  // bf16 elems
constexpr size_t SZ_WG   = (size_t)NT_G * KT * TILE_ELEMS;

// ---- workspace layout (byte offsets, all 16B aligned) ----
constexpr size_t OFF_OUTW  = 0;
constexpr size_t OFF_WIHE  = OFF_OUTW + SZ_OUTW * 2;
constexpr size_t OFF_WHHE  = OFF_WIHE + SZ_WG * 2;
constexpr size_t OFF_WIHD  = OFF_WHHE + SZ_WG * 2;
constexpr size_t OFF_WHHD  = OFF_WIHD + SZ_WG * 2;
constexpr size_t OFF_GI    = OFF_WHHD + SZ_WG * 2;
constexpr size_t OFF_GH    = OFF_GI + (size_t)B * NPAD_G * 4;
constexpr size_t OFF_HST   = OFF_GH + (size_t)B * NPAD_G * 4;
constexpr size_t OFF_HPAD  = OFF_HST + (size_t)B * H * 4;
constexpr size_t OFF_SLOTS = OFF_HPAD + (size_t)B * KP * 2;
constexpr size_t OFF_IDS   = OFF_SLOTS + 64 * 8;

union Frag  { v16bf v; struct { v8bf lo, hi; } p; };
union Pack4 { bf16_t h[4]; unsigned long long u; };

static __device__ __forceinline__
v8f wmma_bf16(v16bf a, v16bf b, v8f c) {
    return __builtin_amdgcn_wmma_f32_16x16x32_bf16(
        /*neg_a=*/false, a, /*neg_b=*/false, b,
        /*c_mod=*/(short)0, c, /*reuse_a=*/false, /*reuse_b=*/false);
}

// A-fragment (16x32 bf16) from an LDS matrix with row stride KP.
// ISA layout: lane L holds M = L%16; K = (e/8)*16 + (L/16)*8 + e%8
// => two contiguous 16-byte chunks per lane.
static __device__ __forceinline__
v16bf load_a(const bf16_t* lds, int mt, int k, int lane) {
    int row = mt * 16 + (lane & 15);
    int c0  = k * 32 + (lane >> 4) * 8;
    Frag f;
    f.p.lo = *(const v8bf*)(lds + (size_t)row * KP + c0);
    f.p.hi = *(const v8bf*)(lds + (size_t)row * KP + c0 + 16);
    return f.v;
}

// fp32 row source -> bf16 LDS matrix [B][KP], zero padded, optional relu.
// rows gathered via tok (or identity when tok == nullptr => hState layout)
static __device__ __forceinline__
void stage_lds(bf16_t* aLds, const int* tok, const float* __restrict__ src,
               int doRelu, int tid, int nthr) {
    constexpr int CQ = KP / 4;                 // 80 quads per row
    for (int i = tid; i < B * CQ; i += nthr) {
        int b = i / CQ, c = (i - b * CQ) * 4;
        float4 v = make_float4(0.f, 0.f, 0.f, 0.f);
        if (c < H) {                            // H % 4 == 0, so quads don't straddle
            size_t row = tok ? (size_t)tok[b] : (size_t)b;
            v = *(const float4*)(src + row * H + c);
            if (doRelu) {
                v.x = fmaxf(v.x, 0.f); v.y = fmaxf(v.y, 0.f);
                v.z = fmaxf(v.z, 0.f); v.w = fmaxf(v.w, 0.f);
            }
        }
        Pack4 p;
        p.h[0] = (bf16_t)v.x; p.h[1] = (bf16_t)v.y;
        p.h[2] = (bf16_t)v.z; p.h[3] = (bf16_t)v.w;
        *(unsigned long long*)(aLds + (size_t)b * KP + c) = p.u;
    }
}

// -------- weight packing: W[N][300] fp32 -> bf16 B-fragment tiles --------
// B (32x16): lane L holds N = L%16, K = (e/8)*16 + (L/16)*8 + e%8  (e = 0..15)
__global__ void pack_weights(const float* __restrict__ W, bf16_t* __restrict__ dst,
                             int N, int nTiles) {
    size_t total = (size_t)nTiles * KT * TILE_ELEMS;
    for (size_t idx = (size_t)blockIdx.x * blockDim.x + threadIdx.x; idx < total;
         idx += (size_t)gridDim.x * blockDim.x) {
        size_t t = idx >> 9;
        int within = (int)(idx & 511);
        int lane = within >> 4, e = within & 15;
        int nT = (int)(t / KT), kT = (int)(t % KT);
        int n = nT * 16 + (lane & 15);
        int k = kT * 32 + (e >> 3) * 16 + (lane >> 4) * 8 + (e & 7);
        float v = (n < N && k < H) ? W[(size_t)n * H + k] : 0.0f;
        dst[idx] = (bf16_t)v;
    }
}

__global__ void init_state(float* hState, bf16_t* hPad, int* ids) {
    int stride = gridDim.x * blockDim.x;
    for (int i = blockIdx.x * blockDim.x + threadIdx.x; i < B * KP; i += stride) {
        hPad[i] = (bf16_t)0.0f;
        if (i < B * H) hState[i] = 0.0f;
        if (i < B)     ids[i] = 0;
    }
}

// -------- GRU matmuls: gi = x@Wih^T, gh = h@Whh^T (biases added later) ------
static __device__ __forceinline__
void mm_gate(const bf16_t* aLds, const bf16_t* __restrict__ wP,
             float* __restrict__ out, int wg, int nWaves, int lane) {
    for (int nT = wg; nT < NT_G; nT += nWaves) {
        v8f acc[4] = {};
        const bf16_t* wbase = wP + (size_t)nT * KT * TILE_ELEMS + (size_t)lane * 16;
#pragma unroll 1      // keep rolled: A-loads depend on runtime k -> no LICM spill
        for (int k = 0; k < KT; ++k) {
            v16bf bfrag = *(const v16bf*)(wbase + (size_t)k * TILE_ELEMS);
#pragma unroll
            for (int mt = 0; mt < 4; ++mt)
                acc[mt] = wmma_bf16(load_a(aLds, mt, k, lane), bfrag, acc[mt]);
        }
        int lh = lane >> 4, nl = lane & 15;
#pragma unroll
        for (int mt = 0; mt < 4; ++mt)
#pragma unroll
            for (int r = 0; r < 8; ++r)
                out[(size_t)(mt * 16 + 8 * lh + r) * NPAD_G + nT * 16 + nl] = acc[mt][r];
    }
}

__global__ void gru_matmul(const int* __restrict__ tok, const float* __restrict__ emb,
                           int doRelu, const float* __restrict__ hState,
                           const bf16_t* __restrict__ wihP, const bf16_t* __restrict__ whhP,
                           float* __restrict__ gi, float* __restrict__ gh) {
    __shared__ bf16_t aLds[B * KP];          // 40 KB
    int tid = threadIdx.x, lane = tid & 31, wave = tid >> 5;
    int wavesPerBlk = blockDim.x >> 5;
    int wg = blockIdx.x * wavesPerBlk + wave;
    int nWaves = gridDim.x * wavesPerBlk;

    stage_lds(aLds, tok, emb, doRelu, tid, blockDim.x);       // A = x
    __syncthreads();
    mm_gate(aLds, wihP, gi, wg, nWaves, lane);
    __syncthreads();
    stage_lds(aLds, nullptr, hState, 0, tid, blockDim.x);     // A = h
    __syncthreads();
    mm_gate(aLds, whhP, gh, wg, nWaves, lane);
}

// -------- GRU gates (elementwise) + new h (fp32 + padded bf16) --------
__global__ void gru_gates(const float* __restrict__ gi, const float* __restrict__ gh,
                          const float* __restrict__ bih, const float* __restrict__ bhh,
                          float* __restrict__ hState, bf16_t* __restrict__ hPad,
                          unsigned long long* __restrict__ slots) {
    int idx = blockIdx.x * blockDim.x + threadIdx.x;
    if (blockIdx.x == 0 && threadIdx.x < 64) slots[threadIdx.x] = 0ull;  // clear argmax slots
    if (idx >= B * H) return;
    int b = idx / H, j = idx - b * H;
    const float* gib = gi + (size_t)b * NPAD_G;
    const float* ghb = gh + (size_t)b * NPAD_G;
    float ir  = gib[j]         + bih[j],         hr = ghb[j]         + bhh[j];
    float iz  = gib[H + j]     + bih[H + j],     hz = ghb[H + j]     + bhh[H + j];
    float inn = gib[2 * H + j] + bih[2 * H + j], hn = ghb[2 * H + j] + bhh[2 * H + j];
    float r = 1.0f / (1.0f + __expf(-(ir + hr)));
    float z = 1.0f / (1.0f + __expf(-(iz + hz)));
    float n = tanhf(inn + r * hn);
    float hNew = (1.0f - z) * n + z * hState[idx];
    hState[idx] = hNew;
    hPad[(size_t)b * KP + j] = (bf16_t)hNew;   // cols 300..319 stay zero (init_state)
}

// -------- argmax emit for one 16-column tile --------
static __device__ __forceinline__
void argmax_tile(const v8f (&acc)[4], int n, float bias, int lane,
                 unsigned long long* sSlots) {
    int lh = lane >> 4, nl = lane & 15;
    unsigned idxKey = ~(unsigned)n;            // ties -> smallest index wins
#pragma unroll
    for (int mt = 0; mt < 4; ++mt) {
#pragma unroll
        for (int r = 0; r < 8; ++r) {
            float logit = acc[mt][r] + bias;
            unsigned u = __float_as_uint(logit);
            unsigned s = (u & 0x80000000u) ? ~u : (u | 0x80000000u);
            unsigned long long key = (((unsigned long long)s) << 32) | idxKey;
#pragma unroll
            for (int d = 8; d >= 1; d >>= 1) { // reduce over 16 cols held by lanes
                unsigned long long o = __shfl_xor(key, d, 16);
                if (o > key) key = o;
            }
            if (nl == 0) atomicMax(&sSlots[mt * 16 + 8 * lh + r], key);
        }
    }
}

// -------- output projection + argmax (log_softmax is argmax-invariant) ------
// 2 vocab n-tiles per wave: B-tiles share A-fragments (2x FLOPs per ds-load)
__global__ void proj_argmax(const bf16_t* __restrict__ hPad, const bf16_t* __restrict__ wP,
                            const float* __restrict__ outB,
                            unsigned long long* __restrict__ gSlots) {
    __shared__ bf16_t hLds[B * KP];                   // 40 KB
    __shared__ unsigned long long sSlots[64];
    int tid = threadIdx.x, lane = tid & 31, wave = tid >> 5;
    if (tid < 64) sSlots[tid] = 0ull;
    for (int i = tid; i < B * KP * 2 / 16; i += blockDim.x)   // copy as uint4
        ((uint4*)hLds)[i] = ((const uint4*)hPad)[i];
    __syncthreads();

    int base = (blockIdx.x * 8 + wave) * 2;            // even, pair-uniform validity
    if (base < NT_V) {
        v8f acc0[4] = {}, acc1[4] = {};
        const bf16_t* w0 = wP + (size_t)base * KT * TILE_ELEMS + (size_t)lane * 16;
        const bf16_t* w1 = w0 + KT * TILE_ELEMS;
#pragma unroll 2
        for (int k = 0; k < KT; ++k) {
            v16bf b0 = *(const v16bf*)(w0 + (size_t)k * TILE_ELEMS);
            v16bf b1 = *(const v16bf*)(w1 + (size_t)k * TILE_ELEMS);
            if (k + 1 < KT) {
                __builtin_prefetch(w0 + (size_t)(k + 1) * TILE_ELEMS, 0, 0);
                __builtin_prefetch(w1 + (size_t)(k + 1) * TILE_ELEMS, 0, 0);
            }
#pragma unroll
            for (int mt = 0; mt < 4; ++mt) {
                v16bf a = load_a(hLds, mt, k, lane);
                acc0[mt] = wmma_bf16(a, b0, acc0[mt]);
                acc1[mt] = wmma_bf16(a, b1, acc1[mt]);
            }
        }
        int nl = lane & 15;
        int n0 = base * 16 + nl;                        // always < V (V % 32 == 0)
        argmax_tile(acc0, n0,      outB[n0],      lane, sSlots);
        argmax_tile(acc1, n0 + 16, outB[n0 + 16], lane, sSlots);
    }
    __syncthreads();
    if (tid < 64 && sSlots[tid])
        atomicMax(&gSlots[tid], sSlots[tid]);
}

__global__ void finalize_step(const unsigned long long* __restrict__ slots,
                              int* __restrict__ ids, float* __restrict__ out, int t) {
    int b = threadIdx.x;
    if (b < B) {
        unsigned n = ~(unsigned)(slots[b] & 0xFFFFFFFFull);
        ids[b] = (int)n;
        out[t * B + b] = (float)n;
    }
}

extern "C" void kernel_launch(void* const* d_in, const int* in_sizes, int n_in,
                              void* d_out, int out_size, void* d_ws, size_t ws_size,
                              hipStream_t stream) {
    (void)in_sizes; (void)n_in; (void)out_size; (void)ws_size;
    const int*   input = (const int*)  d_in[0];
    const float* emb   = (const float*)d_in[1];
    const float* eWih  = (const float*)d_in[2];
    const float* eWhh  = (const float*)d_in[3];
    const float* eBih  = (const float*)d_in[4];
    const float* eBhh  = (const float*)d_in[5];
    const float* dWih  = (const float*)d_in[6];
    const float* dWhh  = (const float*)d_in[7];
    const float* dBih  = (const float*)d_in[8];
    const float* dBhh  = (const float*)d_in[9];
    const float* outW  = (const float*)d_in[10];
    const float* outB  = (const float*)d_in[11];
    float* out = (float*)d_out;

    char* ws = (char*)d_ws;
    bf16_t* pOutW = (bf16_t*)(ws + OFF_OUTW);
    bf16_t* pWihE = (bf16_t*)(ws + OFF_WIHE);
    bf16_t* pWhhE = (bf16_t*)(ws + OFF_WHHE);
    bf16_t* pWihD = (bf16_t*)(ws + OFF_WIHD);
    bf16_t* pWhhD = (bf16_t*)(ws + OFF_WHHD);
    float*  gi     = (float*)(ws + OFF_GI);
    float*  gh     = (float*)(ws + OFF_GH);
    float*  hState = (float*)(ws + OFF_HST);
    bf16_t* hPad   = (bf16_t*)(ws + OFF_HPAD);
    unsigned long long* slots = (unsigned long long*)(ws + OFF_SLOTS);
    int*    ids    = (int*)(ws + OFF_IDS);

    // one-time (per launch) weight repack to bf16 WMMA tiles
    pack_weights<<<2048, 256, 0, stream>>>(outW, pOutW, V, NT_V);
    pack_weights<<<160,  256, 0, stream>>>(eWih, pWihE, H3, NT_G);
    pack_weights<<<160,  256, 0, stream>>>(eWhh, pWhhE, H3, NT_G);
    pack_weights<<<160,  256, 0, stream>>>(dWih, pWihD, H3, NT_G);
    pack_weights<<<160,  256, 0, stream>>>(dWhh, pWhhD, H3, NT_G);
    init_state<<<80, 256, 0, stream>>>(hState, hPad, ids);

    // encoder
    for (int t = 0; t < L; ++t) {
        gru_matmul<<<8, 256, 0, stream>>>(input + t * B, emb, 0, hState,
                                          pWihE, pWhhE, gi, gh);
        gru_gates<<<(B * H) / 256, 256, 0, stream>>>(gi, gh, eBih, eBhh,
                                                     hState, hPad, slots);
    }
    // decoder (greedy)
    for (int t = 0; t < L; ++t) {
        gru_matmul<<<8, 256, 0, stream>>>(ids, emb, 1, hState,
                                          pWihD, pWhhD, gi, gh);
        gru_gates<<<(B * H) / 256, 256, 0, stream>>>(gi, gh, dBih, dBhh,
                                                     hState, hPad, slots);
        proj_argmax<<<(NT_V / 16) + 1, 256, 0, stream>>>(hPad, pOutW, outB, slots);
        finalize_step<<<1, 64, 0, stream>>>(slots, ids, out, t);
    }
}
